// PatchCoreCLIP_70042326663331
// MI455X (gfx1250) — matrix operand order repaired and checked by reading, not compiled, not run on hardware
//
#include <hip/hip_runtime.h>

typedef __bf16 v16bf __attribute__((ext_vector_type(16)));
typedef __bf16 v2bf  __attribute__((ext_vector_type(2)));
typedef float  v8f   __attribute__((ext_vector_type(8)));

#define B_N 1024
#define M_N 262144
#define D_N 512
#define BT  128      // queries per workgroup
#define MT  256      // bank rows per workgroup
#define KT  32       // K chunk (one bf16 WMMA depth)
#define NKC (D_N / KT)
#define NTHREADS 512
#define XSTRIDE 260  // dwords per X row in LDS (256 data + 4 pad -> conflict-free b128)
#define YSTRIDE 20   // dwords per Y row in LDS (16 data + 4 pad  -> conflict-free b128)
#define YBUF    (MT * YSTRIDE)

struct Frag32B { uint4 lo, hi; };

__device__ __forceinline__ unsigned cvt_pk_bf16(float a, float b) {
#if __has_builtin(__builtin_amdgcn_cvt_pk_bf16_f32)
    auto p = __builtin_amdgcn_cvt_pk_bf16_f32(a, b);
    return __builtin_bit_cast(unsigned, p);
#else
    // fptrunc pair + build_vector: backend selects v_cvt_pk_bf16_f32
    v2bf p;
    p[0] = (__bf16)a;
    p[1] = (__bf16)b;
    return __builtin_bit_cast(unsigned, p);
#endif
}

__global__ void patchcore_init_out(unsigned* out_bits) {
    int i = blockIdx.x * blockDim.x + threadIdx.x;
    if (i < B_N) out_bits[i] = 0x7F800000u; // +inf
}

__global__ __launch_bounds__(NTHREADS) void patchcore_min_dist(
        const float* __restrict__ X,        // [B, D] raw image features
        const float* __restrict__ Y,        // [M, D] L2-normalized memory bank
        unsigned*    __restrict__ out_bits) // [B] float bits, pre-set to +inf
{
    __shared__ unsigned sX[BT * XSTRIDE];   // bf16 normalized queries, full D (~133KB)
    __shared__ unsigned sY[2 * YBUF];       // bf16 bank rows, double-buffered (~40KB)
    __shared__ float    sx2[BT];
    __shared__ float    sy2[MT];
    __shared__ unsigned sqmin[BT];

    const int tid  = threadIdx.x;
    const int lane = tid & 31;
    const int w    = tid >> 5;              // wave id 0..15
    const int b0   = blockIdx.x * BT;
    const int m0   = blockIdx.y * MT;

    // ---- stage normalized queries as bf16 in LDS (4 threads per row) ----
    {
        const int r = tid >> 2, s = tid & 3;
        const float4* xrow = (const float4*)(X + (size_t)(b0 + r) * D_N + s * 128);
        float ss = 0.f;
        #pragma unroll
        for (int j = 0; j < 32; ++j) {
            float4 f = xrow[j];
            ss += f.x * f.x + f.y * f.y + f.z * f.z + f.w * f.w;
        }
        ss += __shfl_xor(ss, 1, 32);
        ss += __shfl_xor(ss, 2, 32);        // 4 consecutive lanes share one row
        const float inv = rsqrtf(ss);
        #pragma unroll
        for (int j = 0; j < 32; j += 2) {   // re-load (L0/L2 hit), convert, store
            float4 f0 = xrow[j], f1 = xrow[j + 1];
            uint4 d;
            d.x = cvt_pk_bf16(f0.x * inv, f0.y * inv);
            d.y = cvt_pk_bf16(f0.z * inv, f0.w * inv);
            d.z = cvt_pk_bf16(f1.x * inv, f1.y * inv);
            d.w = cvt_pk_bf16(f1.z * inv, f1.w * inv);
            *(uint4*)&sX[r * XSTRIDE + s * 64 + j * 2] = d;
        }
        if (s == 0) sx2[r] = ss * inv * inv;
        if (tid < BT) sqmin[tid] = 0x7F800000u;
    }

    // two threads share one bank row: thread t handles half (t&1) of row (t>>1)
    const int yr   = tid >> 1;              // 0..255
    const int yh   = tid & 1;               // half-row select (16 floats each)
    const float* yrow = Y + (size_t)(m0 + yr) * D_N + yh * (KT / 2);
    const int ydst = yr * YSTRIDE + yh * 8; // 8 dwords per half-row chunk
    float y2acc = 0.f;

    // ---- prologue: stage K-chunk 0 into buffer 0 ----
    {
        const float4* ysrc = (const float4*)yrow;
        float4 yf[4];
        #pragma unroll
        for (int j = 0; j < 4; ++j) yf[j] = ysrc[j];
        #pragma unroll
        for (int j = 0; j < 4; ++j)
            y2acc += yf[j].x * yf[j].x + yf[j].y * yf[j].y
                   + yf[j].z * yf[j].z + yf[j].w * yf[j].w;
        #pragma unroll
        for (int j = 0; j < 4; j += 2) {
            uint4 d;
            d.x = cvt_pk_bf16(yf[j].x,   yf[j].y);
            d.y = cvt_pk_bf16(yf[j].z,   yf[j].w);
            d.z = cvt_pk_bf16(yf[j+1].x, yf[j+1].y);
            d.w = cvt_pk_bf16(yf[j+1].z, yf[j+1].w);
            *(uint4*)&sY[ydst + j * 2] = d;
        }
    }
    __syncthreads();

    v8f acc[2][4] = {};                      // 32 queries x 64 bank rows per wave
    const int q0    = (w >> 2) * 32;         // wave's query sub-block (0/32/64/96)
    const int moff  = (w & 3) * 64;          // wave's bank-row sub-block
    const int abase = (q0 + (lane & 15)) * XSTRIDE + (lane >> 4) * 4;
    const int bbase = (moff + (lane & 15)) * YSTRIDE + (lane >> 4) * 4;

    for (int kc = 0; kc < NKC; ++kc) {
        const int cur = kc & 1;
        const bool more = (kc + 1 < NKC);

        // 1) issue global loads for next chunk (latency hidden under WMMA)
        float4 yf[4];
        if (more) {
            const float4* ysrc = (const float4*)(yrow + (kc + 1) * KT);
            #pragma unroll
            for (int j = 0; j < 4; ++j) yf[j] = ysrc[j];
            if (kc + 2 < NKC) __builtin_prefetch(yrow + (kc + 2) * KT, 0, 1);
        }

        // 2) A fragments (two 16x32 bf16 tiles) from persistent X
        Frag32B af0, af1;
        const int ak = abase + kc * (KT / 2);
        af0.lo = *(const uint4*)&sX[ak];
        af0.hi = *(const uint4*)&sX[ak + 8];
        af1.lo = *(const uint4*)&sX[ak + 16 * XSTRIDE];
        af1.hi = *(const uint4*)&sX[ak + 16 * XSTRIDE + 8];
        const v16bf a0 = __builtin_bit_cast(v16bf, af0);
        const v16bf a1 = __builtin_bit_cast(v16bf, af1);

        // 3) 4 B fragments x 2 A fragments -> 8 WMMAs
        #pragma unroll
        for (int mt = 0; mt < 4; ++mt) {
            Frag32B bfm;
            const int idx = cur * YBUF + bbase + mt * 16 * YSTRIDE;
            bfm.lo = *(const uint4*)&sY[idx];
            bfm.hi = *(const uint4*)&sY[idx + 8];
            const v16bf b = __builtin_bit_cast(v16bf, bfm);
            acc[0][mt] = __builtin_amdgcn_wmma_f32_16x16x32_bf16(
                            false, a0, false, b, (short)0, acc[0][mt], false, false);
            acc[1][mt] = __builtin_amdgcn_wmma_f32_16x16x32_bf16(
                            false, a1, false, b, (short)0, acc[1][mt], false, false);
        }

        // 4) y2 accumulate (f32) + bf16 pack + store into the other buffer
        if (more) {
            #pragma unroll
            for (int j = 0; j < 4; ++j)
                y2acc += yf[j].x * yf[j].x + yf[j].y * yf[j].y
                       + yf[j].z * yf[j].z + yf[j].w * yf[j].w;
            #pragma unroll
            for (int j = 0; j < 4; j += 2) {
                uint4 d;
                d.x = cvt_pk_bf16(yf[j].x,   yf[j].y);
                d.y = cvt_pk_bf16(yf[j].z,   yf[j].w);
                d.z = cvt_pk_bf16(yf[j+1].x, yf[j+1].y);
                d.w = cvt_pk_bf16(yf[j+1].z, yf[j+1].w);
                *(uint4*)&sY[(1 - cur) * YBUF + ydst + j * 2] = d;
            }
        }
        __syncthreads();
    }

    // combine half-row y2 partials (adjacent lanes) and publish
    y2acc += __shfl_xor(y2acc, 1, 32);
    if (yh == 0) sy2[yr] = y2acc;
    __syncthreads();

    // ---- epilogue: d2 = x2 + y2 - 2*dot, min over bank rows ----
    float xq[2][8], mv[2][8];
    #pragma unroll
    for (int qt = 0; qt < 2; ++qt)
        #pragma unroll
        for (int v = 0; v < 8; ++v) {
            xq[qt][v] = sx2[q0 + qt * 16 + v + 8 * (lane >> 4)];
            mv[qt][v] = 3.4e38f;
        }
    #pragma unroll
    for (int mt = 0; mt < 4; ++mt) {
        const float y2v = sy2[moff + mt * 16 + (lane & 15)];
        #pragma unroll
        for (int qt = 0; qt < 2; ++qt)
            #pragma unroll
            for (int v = 0; v < 8; ++v)
                mv[qt][v] = fminf(mv[qt][v], xq[qt][v] + y2v - 2.0f * acc[qt][mt][v]);
    }
    #pragma unroll
    for (int qt = 0; qt < 2; ++qt)
        #pragma unroll
        for (int v = 0; v < 8; ++v) {        // min across 16 columns (lane halves)
            mv[qt][v] = fminf(mv[qt][v], __shfl_xor(mv[qt][v], 1, 32));
            mv[qt][v] = fminf(mv[qt][v], __shfl_xor(mv[qt][v], 2, 32));
            mv[qt][v] = fminf(mv[qt][v], __shfl_xor(mv[qt][v], 4, 32));
            mv[qt][v] = fminf(mv[qt][v], __shfl_xor(mv[qt][v], 8, 32));
        }
    if ((lane & 15) == 0) {
        #pragma unroll
        for (int qt = 0; qt < 2; ++qt)
            #pragma unroll
            for (int v = 0; v < 8; ++v) {
                const int q = q0 + qt * 16 + v + 8 * (lane >> 4);
                const float dist = sqrtf(fmaxf(mv[qt][v], 1e-12f));
                atomicMin(&sqmin[q], __float_as_uint(dist)); // bits monotone, dist>=0
            }
    }
    __syncthreads();
    if (tid < BT) atomicMin(&out_bits[b0 + tid], sqmin[tid]);
}

extern "C" void kernel_launch(void* const* d_in, const int* in_sizes, int n_in,
                              void* d_out, int out_size, void* d_ws, size_t ws_size,
                              hipStream_t stream) {
    const float* X = (const float*)d_in[0];   // [1024, 512] image_features (f32)
    const float* Y = (const float*)d_in[1];   // [262144, 512] memory_bank (f32)
    unsigned* out_bits = (unsigned*)d_out;    // 1024 f32 outputs, written as bits

    patchcore_init_out<<<(B_N + 255) / 256, 256, 0, stream>>>(out_bits);

    dim3 grid(B_N / BT, M_N / MT);  // (8, 1024): b-tiles fastest -> L2 reuse of bank tile
    patchcore_min_dist<<<grid, NTHREADS, 0, stream>>>(X, Y, out_bits);
}
